// Correlation_23845658427953
// MI455X (gfx1250) — matrix operand order repaired and checked by reading, not compiled, not run on hardware
//
#include <hip/hip_runtime.h>

// Problem constants (from reference): N=8, C=256, H=W=128, MD=4, PATCH=9
#define K_N 8
#define K_C 256
#define K_H 128
#define K_W 128
#define PATCH 9

typedef __attribute__((ext_vector_type(16))) __bf16 v16bf;
typedef __attribute__((ext_vector_type(8)))  __bf16 v8bf;
typedef __attribute__((ext_vector_type(8)))  float  v8f;
typedef __attribute__((ext_vector_type(4)))  float  v4f;

// ---------------------------------------------------------------------------
// Phase 1: NCHW fp32  ->  NHWC bf16 staging (so channel vectors are contiguous
// and WMMA operand fragments are pairs of b128 loads).
// grid: (W/32, C/32, N*H), block: 256 threads.
// ---------------------------------------------------------------------------
__global__ __launch_bounds__(256) void nchw_to_nhwc_bf16(
    const float* __restrict__ src, unsigned short* __restrict__ dst) {
  __shared__ float tile[32][33];
  const int w0 = blockIdx.x * 32;
  const int c0 = blockIdx.y * 32;
  const int nh = blockIdx.z;                 // n*H + h
  const int n  = nh / K_H;
  const int h  = nh % K_H;
  const int t  = threadIdx.x;

  {
    const int cr = t >> 3;                   // 0..31 : channel row in tile
    const int w4 = (t & 7) << 2;             // 0..28 : w offset (float4)
    const float* p = src + (((size_t)n * K_C + (c0 + cr)) * K_H + h) * K_W + w0 + w4;
    v4f v = *(const v4f*)p;                  // coalesced b128 load along w
    tile[cr][w4 + 0] = v.x;
    tile[cr][w4 + 1] = v.y;
    tile[cr][w4 + 2] = v.z;
    tile[cr][w4 + 3] = v.w;
  }
  __syncthreads();
  {
    const int wr = t >> 3;                   // 0..31 : w row
    const int c4 = (t & 7) << 2;             // 0..28 : c offset (4 bf16)
    unsigned short o[4];
#pragma unroll
    for (int j = 0; j < 4; ++j) {
      union { float f; unsigned u; } v;
      v.f = tile[c4 + j][wr];
      // round-to-nearest-even fp32 -> bf16
      unsigned r = v.u + 0x7FFFu + ((v.u >> 16) & 1u);
      o[j] = (unsigned short)(r >> 16);
    }
    unsigned long long pack =
        (unsigned long long)o[0]        | ((unsigned long long)o[1] << 16) |
        ((unsigned long long)o[2] << 32) | ((unsigned long long)o[3] << 48);
    // NHWC: ((n*H + h)*W + w)*C + c   (8-byte aligned b64 store)
    *(unsigned long long*)(dst + ((size_t)nh * K_W + w0 + wr) * K_C + c0 + c4) = pack;
  }
}

// Load one B fragment (lane's 16 contiguous bf16 channels) as two b128 loads.
__device__ __forceinline__ v16bf load_frag(const __bf16* p) {
  v8bf lo = *(const v8bf*)p;
  v8bf hi = *(const v8bf*)(p + 8);
  return __builtin_shufflevector(lo, hi, 0, 1, 2, 3, 4, 5, 6, 7,
                                         8, 9, 10, 11, 12, 13, 14, 15);
}

// ---------------------------------------------------------------------------
// Phase 2: banded-GEMM correlation with v_wmma_f32_16x16x32_bf16.
// One wave handles one (n, h, 16-wide w-tile); A fragments (8 K-steps) are
// held in registers and reused across all 9 dy rows (2 WMMA tiles per dy
// cover columns w0-4 .. w0+27, from which the +/-4 band is extracted).
//
// Zero-padding: D column n depends only on B column n, and the OOB predicate
// is a pure function of the lane's column, so we feed unmasked (address-
// clamped) B fragments to WMMA and zero affected lanes at band extraction.
//
// Addressing: the per-lane column offset is dy-invariant (w-clamp only; a
// row-OOB dy is zeroed at extraction anyway), so per dy only a wave-uniform
// row base changes -> loads lower to saddr+voffset with no per-dy VALU math.
// B fragments are double-buffered so one load clause is always in flight
// ahead of the WMMAs consuming the previous K-step.
// ---------------------------------------------------------------------------
__global__ __launch_bounds__(256) void corr_wmma_kernel(
    const unsigned short* __restrict__ t1,
    const unsigned short* __restrict__ t2,
    float* __restrict__ out) {
  constexpr int WPB = 8;                       // waves per block
  __shared__ float res[WPB][PATCH][16];        // per-wave result staging

  const int wave = threadIdx.x >> 5;
  const int lane = threadIdx.x & 31;
  const int task = blockIdx.x * WPB + wave;    // 0 .. N*H*(W/16)-1 = 8191
  const int wt   = task & 7;                   // w tile (W/16 = 8)
  const int h    = (task >> 3) & (K_H - 1);
  const int n    = task >> 10;
  const int w0   = wt << 4;

  const int half = lane >> 4;                  // K sub-block selector
  const int col  = lane & 15;                  // row (A) / column (B) index

  const __bf16* T1 = (const __bf16*)t1;
  const __bf16* T2 = (const __bf16*)t2;

  // ---- A operand: 16x32 bf16 fragments for all 8 K-steps (kept in VGPRs) ---
  // lane(m, half) holds K = {16*half+0..15} of pixel w0+m.
  v16bf A[8];
  {
    const __bf16* ab = T1 + ((size_t)(n * K_H + h) * K_W + (w0 + col)) * K_C;
#pragma unroll
    for (int k = 0; k < 8; ++k) {
      const int c0 = k * 32;
      v8bf lo = *(const v8bf*)(ab + c0 + 16 * half);
      v8bf hi = *(const v8bf*)(ab + c0 + 16 * half + 8);
      A[k] = __builtin_shufflevector(lo, hi, 0, 1, 2, 3, 4, 5, 6, 7,
                                             8, 9, 10, 11, 12, 13, 14, 15);
    }
  }

  // dy-invariant per-lane column offsets (w-clamped; elements, fits 32-bit).
  const int  w1   = w0 - 4 + col;              // tile 1 columns: w0-4 .. w0+11
  const int  w2   = w0 + 12 + col;             // tile 2 columns: w0+12 .. w0+27
  const bool wok1 = (w1 >= 0) && (w1 < K_W);
  const bool wok2 = (w2 < K_W);
  const int  off1 = (wok1 ? w1 : 0) * K_C + 16 * half;
  const int  off2 = (wok2 ? w2 : 0) * K_C + 16 * half;

  for (int dy = 0; dy < PATCH; ++dy) {
    const int  hh  = h + dy - 4;
    const bool hok = (hh >= 0) && (hh < K_H);
    const int  hc  = hok ? hh : 0;
    const bool ok1 = hok && wok1;
    const bool ok2 = hok && wok2;

    // wave-uniform row base (SGPR); per-lane part is off1/off2 (voffset).
    const __bf16* rowb = T2 + (size_t)(n * K_H + hc) * (K_W * K_C);

    v16bf B1 = load_frag(rowb + off1);
    v16bf B2 = load_frag(rowb + off2);
    v8f acc1 = {};
    v8f acc2 = {};
#pragma unroll
    for (int k = 0; k < 8; ++k) {
      v16bf nB1, nB2;
      if (k < 7) {                             // issue next K-step's loads
        nB1 = load_frag(rowb + off1 + (k + 1) * 32);
        nB2 = load_frag(rowb + off2 + (k + 1) * 32);
      }
      acc1 = __builtin_amdgcn_wmma_f32_16x16x32_bf16(
          false, A[k], false, B1, (short)0, acc1, false, false);
      acc2 = __builtin_amdgcn_wmma_f32_16x16x32_bf16(
          false, A[k], false, B2, (short)0, acc2, false, false);
      if (k < 7) {
        B1 = nB1;
        B2 = nB2;
      }
    }

    // Extract the +/-4 band: D element (VGPR r, lane) = G[m, col-of-tile]
    // with m = r + 8*half.  dx = (w' - (w0+m)) + 4.  OOB columns -> 0.
#pragma unroll
    for (int r = 0; r < 8; ++r) {
      const int m   = r + 8 * half;
      const int dx1 = col - m;                 // tile 1: w' = w0-4+col
      if (dx1 >= 0 && dx1 < PATCH) res[wave][dx1][m] = ok1 ? acc1[r] : 0.0f;
      const int dx2 = col + 16 - m;            // tile 2: w' = w0+12+col
      if (dx2 >= 0 && dx2 < PATCH) res[wave][dx2][m] = ok2 ? acc2[r] : 0.0f;
    }

    // DS ops are in-order within a wave; compiler inserts s_wait_dscnt for
    // the VGPR dependency of the reads below.
    const size_t obase =
        ((((size_t)n * PATCH + dy) * PATCH) * K_H + h) * (size_t)K_W + w0;
#pragma unroll
    for (int i = 0; i < 5; ++i) {
      const int idx = lane + i * 32;
      if (idx < PATCH * 16) {
        const int dx = idx >> 4;
        const int m  = idx & 15;
        out[obase + (size_t)dx * (K_H * K_W) + m] = res[wave][dx][m];
      }
    }
  }
}

// ---------------------------------------------------------------------------
extern "C" void kernel_launch(void* const* d_in, const int* in_sizes, int n_in,
                              void* d_out, int out_size, void* d_ws, size_t ws_size,
                              hipStream_t stream) {
  const float* in1 = (const float*)d_in[0];
  const float* in2 = (const float*)d_in[1];

  // Workspace: two NHWC bf16 staging buffers, 67 MB each (134 MB total).
  unsigned short* T1 = (unsigned short*)d_ws;
  unsigned short* T2 = T1 + (size_t)K_N * K_H * K_W * K_C;

  dim3 tgrid(K_W / 32, K_C / 32, K_N * K_H);   // (4, 8, 1024)
  nchw_to_nhwc_bf16<<<tgrid, 256, 0, stream>>>(in1, T1);
  nchw_to_nhwc_bf16<<<tgrid, 256, 0, stream>>>(in2, T2);

  // 8192 wave-tasks (n*h*wtile), 8 waves (256 threads) per block.
  const int blocks = (K_N * K_H * (K_W / 16)) / 8;  // 1024
  corr_wmma_kernel<<<blocks, 256, 0, stream>>>(T1, T2, (float*)d_out);
}
